// Discriminator_53008486367874
// MI455X (gfx1250) — compile-verified, matrix operand order checked by reading, not run on hardware
//
#include <hip/hip_runtime.h>

// ---------------------------------------------------------------------------
// CDNA5 (gfx1250) wave32 WMMA implementation of the discriminator forward.
// All matmuls run on v_wmma_f32_16x16x32_f16 (f16 in, f32 accumulate).
// ---------------------------------------------------------------------------

typedef __attribute__((ext_vector_type(16))) _Float16 v16h;
typedef __attribute__((ext_vector_type(8)))  float    v8f;

#define BB 512
#define TT 256
#define FF 64
#define HH 256
#define CF 128
#define EPS 1e-3f

union FragU { v16h v; float4 q[2]; };

// Load one 16x32 f16 A/B fragment from a row-major [*, ldk] matrix.
// Per ISA 7.12.2 (16-bit A 16x32): lanes 0-15 hold K={0..7,16..23},
// lanes 16-31 hold K={8..15,24..31}; B mirrors with N in place of M.
__device__ __forceinline__ v16h load_frag(const _Float16* rowk, int hg) {
    FragU u;
    u.q[0] = *reinterpret_cast<const float4*>(rowk + hg * 8);
    u.q[1] = *reinterpret_cast<const float4*>(rowk + 16 + hg * 8);
    return u.v;
}

// Non-hoistable fragment gather for 4 B-tiles at a common k offset.
// pN already includes the lane's row and the hg*8 element shift.
// BOFF = k0*2 bytes (compile-time constant via full unroll).
__device__ __forceinline__ void gload_frag4(FragU& f0, FragU& f1, FragU& f2, FragU& f3,
                                            const _Float16* p0, const _Float16* p1,
                                            const _Float16* p2, const _Float16* p3,
                                            int boff_lo, int boff_hi) {
    asm volatile(
        "global_load_b128 %0, %8, off offset:%c12\n\t"
        "global_load_b128 %1, %8, off offset:%c13\n\t"
        "global_load_b128 %2, %9, off offset:%c12\n\t"
        "global_load_b128 %3, %9, off offset:%c13\n\t"
        "global_load_b128 %4, %10, off offset:%c12\n\t"
        "global_load_b128 %5, %10, off offset:%c13\n\t"
        "global_load_b128 %6, %11, off offset:%c12\n\t"
        "global_load_b128 %7, %11, off offset:%c13\n\t"
        "s_wait_loadcnt 0x0"
        : "=v"(f0.q[0]), "=v"(f0.q[1]), "=v"(f1.q[0]), "=v"(f1.q[1]),
          "=v"(f2.q[0]), "=v"(f2.q[1]), "=v"(f3.q[0]), "=v"(f3.q[1])
        : "v"(p0), "v"(p1), "v"(p2), "v"(p3), "i"(boff_lo), "i"(boff_hi)
        : "memory");
}

__device__ __forceinline__ v8f wmma16(v16h a, v16h b, v8f c) {
    return __builtin_amdgcn_wmma_f32_16x16x32_f16(false, a, false, b,
                                                  (short)0, c, false, false);
}

__device__ __forceinline__ float lrelu(float x, float s) {
    return x >= 0.f ? x : s * x;
}
__device__ __forceinline__ float sigmoidf(float x) {
    return 1.f / (1.f + __expf(-x));
}

// ---------------------------------------------------------------------------
// Prep kernels (run once per launch; tiny relative to the GEMMs)
// ---------------------------------------------------------------------------
__global__ __launch_bounds__(256) void cvt_f16_kernel(const float* __restrict__ src,
                                                      _Float16* __restrict__ dst, int n) {
    int i = blockIdx.x * 256 + threadIdx.x;
    if (i < n) dst[i] = (_Float16)src[i];
}

// src: f32 [K][N] -> dst: f16 [N][K]
__global__ __launch_bounds__(256) void transpose_f16_kernel(const float* __restrict__ src,
                                                            _Float16* __restrict__ dst,
                                                            int K, int N) {
    int i = blockIdx.x * 256 + threadIdx.x;
    if (i >= K * N) return;
    int k = i / N, n = i - k * N;
    dst[(size_t)n * K + k] = (_Float16)src[i];
}

// Fused affine for BN1->BN3 (cols 0..255) and BN2->BN3 (cols 256..511):
// bn(x)=s*x+t with s=g*rsqrt(v+eps), t=b-m*s;  bn3(bnk(x)) = (s3*sk)x + s3*tk+t3
__global__ __launch_bounds__(256) void eff_kernel(
    const float* g1, const float* b1, const float* m1, const float* v1,
    const float* g2, const float* b2, const float* m2, const float* v2,
    const float* g3, const float* b3, const float* m3, const float* v3,
    float* __restrict__ effs, float* __restrict__ efft) {
    int j = blockIdx.x * 256 + threadIdx.x;
    if (j >= 512) return;
    float s3 = g3[j] * rsqrtf(v3[j] + EPS);
    float t3 = b3[j] - m3[j] * s3;
    float sk, tk;
    if (j < 256) {
        float s = g1[j] * rsqrtf(v1[j] + EPS);
        sk = s; tk = b1[j] - m1[j] * s;
    } else {
        int h = j - 256;
        float s = g2[h] * rsqrtf(v2[h] + EPS);
        sk = s; tk = b2[h] - m2[h] * s;
    }
    effs[j] = s3 * sk;
    efft[j] = s3 * tk + t3;
}

// ---------------------------------------------------------------------------
// LSTM: one workgroup (512 threads = 16 waves) owns 16 batch rows for all
// 256 steps. Wave w owns gate column-group w (16 cols of H): 4 z-tiles
// (i,f,g,o). Weight B-fragments are re-fetched every step via asm-volatile
// global_load_b128 (L2-hot, un-hoistable -> no register spills). h is
// double-buffered in LDS; c lives in D-fragment-layout registers.
// Epilogue: lrelu(0.2) -> BN1 -> BN3 fused -> A0[:, 0:256] (f16).
// ---------------------------------------------------------------------------
__global__ __launch_bounds__(512) void lstm_kernel(
    const _Float16* __restrict__ xh,   // [B][T][F] f16
    const _Float16* __restrict__ WxT,  // [4H][F]  f16
    const _Float16* __restrict__ WhT,  // [4H][H]  f16
    const float*    __restrict__ bL,   // [4H]
    const float*    __restrict__ effs,
    const float*    __restrict__ efft,
    _Float16*       __restrict__ A0)   // [B][2H] f16, cols 0..255
{
    __shared__ _Float16 hsh[2][16][HH];

    const int tid = threadIdx.x;
    const int lane = tid & 31, w = tid >> 5;         // 16 waves
    const int l15 = lane & 15, hg = lane >> 4;
    const int r0 = blockIdx.x * 16;

    for (int i = tid; i < 16 * HH; i += 512) ((_Float16*)hsh[0])[i] = (_Float16)0.f;
    __syncthreads();

    // per-lane B row pointers for the 4 gate tiles (include hg*8 shift)
    const _Float16* pbh[4];
    const _Float16* pbx[4];
    float bias[4];
#pragma unroll
    for (int g = 0; g < 4; ++g) {
        int n = g * HH + w * 16 + l15;
        pbh[g] = WhT + (size_t)n * HH + hg * 8;
        pbx[g] = WxT + (size_t)n * FF + hg * 8;
        bias[g] = bL[n];
    }

    float cst[8];
#pragma unroll
    for (int r = 0; r < 8; ++r) cst[r] = 0.f;

    for (int t = 0; t < TT; ++t) {
        const int cur = t & 1, nxt = cur ^ 1;
        v8f acc[4];
#pragma unroll
        for (int g = 0; g < 4; ++g)
            acc[g] = (v8f){0.f, 0.f, 0.f, 0.f, 0.f, 0.f, 0.f, 0.f};

        // x_t @ Wx  (K = 64, 2 k-steps)
        const _Float16* xrow = xh + ((size_t)(r0 + l15) * TT + t) * FF;
#pragma unroll
        for (int ks = 0; ks < 2; ++ks) {
            v16h a = load_frag(xrow + ks * 32, hg);
            FragU b0, b1, b2, b3;
            gload_frag4(b0, b1, b2, b3, pbx[0], pbx[1], pbx[2], pbx[3],
                        ks * 64, ks * 64 + 32);
            acc[0] = wmma16(a, b0.v, acc[0]);
            acc[1] = wmma16(a, b1.v, acc[1]);
            acc[2] = wmma16(a, b2.v, acc[2]);
            acc[3] = wmma16(a, b3.v, acc[3]);
        }
        // h @ Wh  (K = 256, 8 k-steps); A from LDS (ds_load_b128)
#pragma unroll
        for (int ks = 0; ks < 8; ++ks) {
            v16h a = load_frag(&hsh[cur][l15][ks * 32], hg);
            FragU b0, b1, b2, b3;
            gload_frag4(b0, b1, b2, b3, pbh[0], pbh[1], pbh[2], pbh[3],
                        ks * 64, ks * 64 + 32);
            acc[0] = wmma16(a, b0.v, acc[0]);
            acc[1] = wmma16(a, b1.v, acc[1]);
            acc[2] = wmma16(a, b2.v, acc[2]);
            acc[3] = wmma16(a, b3.v, acc[3]);
        }
        // gates: i,f,g,o ; c = sig(f)*c + sig(i)*tanh(g); h = sig(o)*tanh(c)
        {
            int col = w * 16 + l15;
#pragma unroll
            for (int r = 0; r < 8; ++r) {
                float zi = acc[0][r] + bias[0];
                float zf = acc[1][r] + bias[1];
                float zg = acc[2][r] + bias[2];
                float zo = acc[3][r] + bias[3];
                float c = sigmoidf(zf) * cst[r] + sigmoidf(zi) * tanhf(zg);
                cst[r] = c;
                hsh[nxt][r + 8 * hg][col] = (_Float16)(sigmoidf(zo) * tanhf(c));
            }
        }
        __syncthreads();
    }

    // final h is in hsh[0] (t=255 wrote buffer 0): lrelu 0.2 + fused BN1*BN3
    {
        int row = tid >> 5;            // 0..15
        int c0 = (tid & 31) * 8;       // 8 cols per thread
#pragma unroll
        for (int j = 0; j < 8; ++j) {
            int col = c0 + j;
            float v = (float)hsh[0][row][col];
            v = lrelu(v, 0.2f);
            v = effs[col] * v + efft[col];
            A0[(size_t)(r0 + row) * (2 * HH) + col] = (_Float16)v;
        }
    }
}

// ---------------------------------------------------------------------------
// Generic WMMA GEMM: Out = post( A[MxK] @ BT[NxK]^T + bias )
// post: lrelu(slope) then optional per-column affine (effs/efft, fused BN).
// One wave per 16x16 output tile, 8 waves per block.
// ---------------------------------------------------------------------------
__global__ __launch_bounds__(256) void gemm_wmma_kernel(
    const _Float16* __restrict__ A, const _Float16* __restrict__ BT,
    const float* __restrict__ bias, _Float16* __restrict__ Out,
    int M, int N, int K, int ldout, int ncol0, float slope,
    const float* __restrict__ effs, const float* __restrict__ efft)
{
    int gw = (blockIdx.x * 256 + threadIdx.x) >> 5;
    int lane = threadIdx.x & 31, l15 = lane & 15, hg = lane >> 4;
    int ntiles = N >> 4;
    int mt = gw / ntiles, nt = gw - mt * ntiles;
    if (mt >= (M >> 4)) return;  // wave-uniform

    const _Float16* arow = A + (size_t)(mt * 16 + l15) * K;
    const _Float16* brow = BT + (size_t)(nt * 16 + l15) * K;

    v8f acc = (v8f){0.f, 0.f, 0.f, 0.f, 0.f, 0.f, 0.f, 0.f};
#pragma unroll 4
    for (int kk = 0; kk < K; kk += 32) {
        v16h a = load_frag(arow + kk, hg);
        v16h b = load_frag(brow + kk, hg);
        acc = wmma16(a, b, acc);
    }

    int n = nt * 16 + l15;
    float bv = bias ? bias[n] : 0.f;
    float es = effs ? effs[ncol0 + n] : 1.f;
    float et = effs ? efft[ncol0 + n] : 0.f;
#pragma unroll
    for (int r = 0; r < 8; ++r) {
        float v = acc[r] + bv;
        v = lrelu(v, slope);
        v = es * v + et;
        Out[(size_t)(mt * 16 + r + 8 * hg) * ldout + ncol0 + n] = (_Float16)v;
    }
}

// ---------------------------------------------------------------------------
// Final layer: out[b] = sigmoid( A2[b,:] . W3 + b3 ), one wave per row.
// ---------------------------------------------------------------------------
__global__ __launch_bounds__(256) void final_kernel(
    const _Float16* __restrict__ A2, const float* __restrict__ W3,
    const float* __restrict__ b3, float* __restrict__ out, int M, int K)
{
    int row = (blockIdx.x * 256 + threadIdx.x) >> 5;
    int lane = threadIdx.x & 31;
    if (row >= M) return;
    const _Float16* a = A2 + (size_t)row * K;
    float s = 0.f;
    for (int k = lane; k < K; k += 32) s += (float)a[k] * W3[k];
#pragma unroll
    for (int off = 16; off > 0; off >>= 1) s += __shfl_down(s, off, 32);
    if (lane == 0) out[row] = 1.f / (1.f + __expf(-(s + b3[0])));
}

// ---------------------------------------------------------------------------
// Host launcher
// ---------------------------------------------------------------------------
extern "C" void kernel_launch(void* const* d_in, const int* in_sizes, int n_in,
                              void* d_out, int out_size, void* d_ws, size_t ws_size,
                              hipStream_t stream) {
    const float* x     = (const float*)d_in[0];   // (512,256,64)
    const float* Wx    = (const float*)d_in[1];   // (64,1024)
    const float* Wh    = (const float*)d_in[2];   // (256,1024)
    const float* bL    = (const float*)d_in[3];   // (1024)
    const float* bn1g  = (const float*)d_in[4];
    const float* bn1b  = (const float*)d_in[5];
    const float* bn1m  = (const float*)d_in[6];
    const float* bn1v  = (const float*)d_in[7];
    const float* Wc    = (const float*)d_in[8];   // (64,128)
    const float* bc    = (const float*)d_in[9];   // (128)
    const float* Wd    = (const float*)d_in[10];  // (32768,256)
    const float* bd    = (const float*)d_in[11];  // (256)
    const float* bn2g  = (const float*)d_in[12];
    const float* bn2b  = (const float*)d_in[13];
    const float* bn2m  = (const float*)d_in[14];
    const float* bn2v  = (const float*)d_in[15];
    const float* bn3g  = (const float*)d_in[16];
    const float* bn3b  = (const float*)d_in[17];
    const float* bn3m  = (const float*)d_in[18];
    const float* bn3v  = (const float*)d_in[19];
    const float* W1    = (const float*)d_in[20];  // (512,1024)
    const float* b1    = (const float*)d_in[21];
    const float* W2    = (const float*)d_in[22];  // (1024,1024)
    const float* b2    = (const float*)d_in[23];
    const float* W3    = (const float*)d_in[24];  // (1024,1)
    const float* b3    = (const float*)d_in[25];
    float* out = (float*)d_out;

    char* wp = (char*)d_ws;
    auto alloc = [&](size_t bytes) {
        char* p = wp;
        wp += (bytes + 255) & ~(size_t)255;
        return p;
    };
    _Float16* xh   = (_Float16*)alloc((size_t)BB * TT * FF * 2);        // 16.8 MB
    _Float16* WxT  = (_Float16*)alloc((size_t)4 * HH * FF * 2);         // 128 KB
    _Float16* WhT  = (_Float16*)alloc((size_t)4 * HH * HH * 2);         // 512 KB
    _Float16* WcT  = (_Float16*)alloc((size_t)CF * FF * 2);             // 16 KB
    _Float16* WdT  = (_Float16*)alloc((size_t)HH * TT * CF * 2);        // 16.8 MB
    _Float16* W1T  = (_Float16*)alloc((size_t)1024 * 512 * 2);          // 1 MB
    _Float16* W2T  = (_Float16*)alloc((size_t)1024 * 1024 * 2);         // 2 MB
    _Float16* C1   = (_Float16*)alloc((size_t)BB * TT * CF * 2);        // 33.6 MB
    _Float16* A0   = (_Float16*)alloc((size_t)BB * 2 * HH * 2);         // 512 KB
    _Float16* A1   = (_Float16*)alloc((size_t)BB * 1024 * 2);           // 1 MB
    _Float16* A2   = (_Float16*)alloc((size_t)BB * 1024 * 2);           // 1 MB
    float*    effs = (float*)alloc(512 * 4);
    float*    efft = (float*)alloc(512 * 4);

    auto cdiv = [](int a, int b) { return (a + b - 1) / b; };

    // --- prep: x -> f16, weights -> f16 transposed, fused BN affines ---
    {
        int n = BB * TT * FF;
        cvt_f16_kernel<<<cdiv(n, 256), 256, 0, stream>>>(x, xh, n);
    }
    transpose_f16_kernel<<<cdiv(FF * 4 * HH, 256), 256, 0, stream>>>(Wx, WxT, FF, 4 * HH);
    transpose_f16_kernel<<<cdiv(HH * 4 * HH, 256), 256, 0, stream>>>(Wh, WhT, HH, 4 * HH);
    transpose_f16_kernel<<<cdiv(FF * CF, 256), 256, 0, stream>>>(Wc, WcT, FF, CF);
    transpose_f16_kernel<<<cdiv(TT * CF * HH, 256), 256, 0, stream>>>(Wd, WdT, TT * CF, HH);
    transpose_f16_kernel<<<cdiv(512 * 1024, 256), 256, 0, stream>>>(W1, W1T, 512, 1024);
    transpose_f16_kernel<<<cdiv(1024 * 1024, 256), 256, 0, stream>>>(W2, W2T, 1024, 1024);
    eff_kernel<<<2, 256, 0, stream>>>(bn1g, bn1b, bn1m, bn1v,
                                      bn2g, bn2b, bn2m, bn2v,
                                      bn3g, bn3b, bn3m, bn3v, effs, efft);

    // --- branch A: LSTM + lrelu(0.2) + BN1 (+BN3) -> A0[:, 0:256] ---
    lstm_kernel<<<BB / 16, 512, 0, stream>>>(xh, WxT, WhT, bL, effs, efft, A0);

    // --- branch B: conv1x1 + lrelu(0.2) -> C1 (f16) ---
    {
        int M = BB * TT, N = CF;
        int waves = (M / 16) * (N / 16);
        gemm_wmma_kernel<<<cdiv(waves, 8), 256, 0, stream>>>(
            xh, WcT, bc, C1, M, N, FF, CF, 0, 0.2f, nullptr, nullptr);
    }
    // flatten @ Wd + lrelu(0.2) + BN2 (+BN3) -> A0[:, 256:512]
    {
        int M = BB, N = HH, K = TT * CF;
        int waves = (M / 16) * (N / 16);
        gemm_wmma_kernel<<<cdiv(waves, 8), 256, 0, stream>>>(
            C1, WdT, bd, A0, M, N, K, 2 * HH, HH, 0.2f, effs, efft);
    }
    // --- head: A0 @ W1 -> lrelu(0.3) -> A1 ; A1 @ W2 -> lrelu(0.3) -> A2 ---
    {
        int M = BB, N = 1024, K = 512;
        int waves = (M / 16) * (N / 16);
        gemm_wmma_kernel<<<cdiv(waves, 8), 256, 0, stream>>>(
            A0, W1T, b1, A1, M, N, K, 1024, 0, 0.3f, nullptr, nullptr);
    }
    {
        int M = BB, N = 1024, K = 1024;
        int waves = (M / 16) * (N / 16);
        gemm_wmma_kernel<<<cdiv(waves, 8), 256, 0, stream>>>(
            A1, W2T, b2, A2, M, N, K, 1024, 0, 0.3f, nullptr, nullptr);
    }
    // --- final: sigmoid(A2 @ W3 + b3) -> out (512 floats) ---
    final_kernel<<<cdiv(BB * 32, 256), 256, 0, stream>>>(A2, W3, b3, out, BB, 1024);

    (void)in_sizes; (void)n_in; (void)out_size; (void)ws_size;
}
